// Attention_hw_16234976379353
// MI455X (gfx1250) — matrix operand order repaired and checked by reading, not compiled, not run on hardware
//
#include <hip/hip_runtime.h>

#define DIM   192
#define C3    576
#define HEADS 6
#define DH    32
#define WS    8
#define TOK   64
#define TILE  10
#define NPIX  100
#define NPAD  112
#define IMG   256

typedef float  v8f   __attribute__((ext_vector_type(8)));
typedef __bf16 v16bf __attribute__((ext_vector_type(16)));
typedef __bf16 v8bf  __attribute__((ext_vector_type(8)));

#if __has_builtin(__builtin_amdgcn_global_load_async_to_lds_b32)
#define HAVE_ASYNC_LDS 1
typedef __attribute__((address_space(1))) int gint;   // global (AS1) dword
typedef __attribute__((address_space(3))) int lint;   // LDS (AS3) dword
#endif

// ---------------- WMMA fragment helpers (CDNA5 ISA 7.12.2 layouts) ----------------

// A-matrix 16x32 bf16 from GLOBAL bf16, row-major [M][K]; two global_load_b128 per lane.
// lane<16: elems 0..7 = K 0..7, 8..15 = K 16..23 ; lane>=16: K 8..15 / 24..31.
__device__ __forceinline__ v16bf a_frag_global_bf16(const __bf16* __restrict__ W,
                                                    int ldw, int mbase, int kbase) {
  int l = threadIdx.x & 31;
  const __bf16* row = W + (size_t)(mbase + (l & 15)) * ldw + kbase + ((l & 16) ? 8 : 0);
  v8bf lo = *(const v8bf*)row;
  v8bf hi = *(const v8bf*)(row + 16);
  v16bf a;
#pragma unroll
  for (int i = 0; i < 8; ++i) { a[i] = lo[i]; a[8 + i] = hi[i]; }
  return a;
}

// A-matrix 16x32 bf16 from LDS, row-major [M][K]; 16B-aligned ds_load_b128 pairs.
__device__ __forceinline__ v16bf a_frag_lds(const __bf16* A, int lda, int mbase, int kbase) {
  int l = threadIdx.x & 31;
  const __bf16* row = A + (mbase + (l & 15)) * lda + kbase + ((l & 16) ? 8 : 0);
  v8bf lo = *(const v8bf*)row;
  v8bf hi = *(const v8bf*)(row + 16);
  v16bf a;
#pragma unroll
  for (int i = 0; i < 8; ++i) { a[i] = lo[i]; a[8 + i] = hi[i]; }
  return a;
}

// B-matrix 32x16 bf16 from LDS stored TRANSPOSED (N-major): Bt[n*ldt + k] == B[k][n].
// lane<16: elems 0..7 = K 0..7, 8..15 = K 16..23 ; lane>=16: K 8..15 / 24..31 (ds_load_b128 pairs).
__device__ __forceinline__ v16bf b_frag_ldsT(const __bf16* Bt, int ldt, int kbase, int nbase) {
  int l = threadIdx.x & 31;
  const __bf16* src = Bt + (nbase + (l & 15)) * ldt + kbase + ((l & 16) ? 8 : 0);
  v8bf lo = *(const v8bf*)src;
  v8bf hi = *(const v8bf*)(src + 16);
  v16bf b;
#pragma unroll
  for (int i = 0; i < 8; ++i) { b[i] = lo[i]; b[8 + i] = hi[i]; }
  return b;
}

__device__ __forceinline__ v8f wmma_bf16(v16bf a, v16bf b, v8f c) {
  return __builtin_amdgcn_wmma_f32_16x16x32_bf16(false, a, false, b, (short)0, c, false, false);
}

// ---------------- prep: convert w1 (576x192) and wp (192x192) to bf16 in d_ws ----------------
__global__ void cvt_weights(const float* __restrict__ w1, const float* __restrict__ wp,
                            __bf16* __restrict__ ws) {
  int idx = blockIdx.x * 256 + threadIdx.x;
  const int n1 = C3 * DIM;
  const int n2 = DIM * DIM;
  if (idx < n1)            ws[idx]      = (__bf16)w1[idx];
  else if (idx < n1 + n2)  ws[idx]      = (__bf16)wp[idx - n1];
}

// ---------------- fused kernel: one workgroup per 8x8 window ----------------
// LDS (dynamic, 245760 B total):
//   xs    [NPAD][DIM] bf16  @ 0       (43008)   x halo tile, N-major for B frags
//   xstg  [DIM][NPIX] f32   @ 43008   (76800)   async-load staging (overlays ys)
//   ys    [C3][NPAD]  bf16  @ 43008   (129024)  conv1 output (with halo)
//   qs    [TOK][DIM]  bf16  @ 172032  (24576)   q (A-operand layout)
//   ks    [TOK][DIM]  bf16  @ 196608  (24576)   k (Bt layout for QK^T)
//   vs    [DIM][TOK]  bf16  @ 221184  (24576)   v (Bt layout for AV)
//   wdl/bdl f32             @ 0       (23040)   dw weights (phase 2 only, xs dead)
//   probs [8][16][64] bf16  @ 0       (16384)   per-wave softmax scratch
//   ao    [TOK][DIM]  bf16  @ 16384   (24576)   attn out (Bt layout for proj)
__global__ __launch_bounds__(256, 1)
void fused_window_attn(const float* __restrict__ x,
                       const __bf16* __restrict__ w1bf, const float* __restrict__ b1,
                       const float* __restrict__ wd,    const float* __restrict__ bd,
                       const __bf16* __restrict__ wpbf, const float* __restrict__ bp,
                       const float* __restrict__ temp,  float* __restrict__ out) {
  extern __shared__ char smem[];
  __bf16* xs    = (__bf16*)(smem);
  __bf16* ys    = (__bf16*)(smem + 43008);
  __bf16* qs    = (__bf16*)(smem + 172032);
  __bf16* ks    = (__bf16*)(smem + 196608);
  __bf16* vs    = (__bf16*)(smem + 221184);
  __bf16* probs = (__bf16*)(smem);
  __bf16* ao    = (__bf16*)(smem + 16384);

  const int tid   = threadIdx.x;
  const int wave  = tid >> 5;
  const int lane  = tid & 31;
  const int half8 = (lane & 16) ? 8 : 0;
  const int ln    = lane & 15;

  const int blk = blockIdx.x;
  const int bi  = blk >> 10;
  const int wi  = (blk >> 5) & 31;
  const int wj  = blk & 31;
  const int r0  = wi * WS, c0 = wj * WS;

  // ---- phase 0: x halo tile -> LDS ----
#ifdef HAVE_ASYNC_LDS
  {
    gint* gb = (gint*)x;                    // dword-indexed global pointer
    lint* lb = (lint*)(smem + 43008);       // dword-indexed LDS staging
    for (int idx = tid; idx < DIM * NPIX; idx += 256) {
      int c = idx / NPIX, p = idx - c * NPIX;
      int ly = p / TILE, lx = p - ly * TILE;
      int gy = r0 - 1 + ly, gx = c0 - 1 + lx;
      if ((unsigned)gy < IMG && (unsigned)gx < IMG)
        __builtin_amdgcn_global_load_async_to_lds_b32(
            gb + (((size_t)(bi * DIM + c) << 16) + (gy << 8) + gx),
            lb + idx, 0, 0);
    }
    asm volatile("s_wait_asynccnt 0x0" ::: "memory");
  }
  __syncthreads();
  {
    const float* xstg = (const float*)(smem + 43008);
    for (int idx = tid; idx < DIM * NPAD; idx += 256) {
      int c = idx / NPAD, p = idx - c * NPAD;
      float v = 0.f;
      if (p < NPIX) {
        int ly = p / TILE, lx = p - ly * TILE;
        int gy = r0 - 1 + ly, gx = c0 - 1 + lx;
        if ((unsigned)gy < IMG && (unsigned)gx < IMG) v = xstg[c * NPIX + p];
      }
      xs[p * DIM + c] = (__bf16)v;
    }
  }
#else
  for (int idx = tid; idx < DIM * NPAD; idx += 256) {
    int c = idx / NPAD, p = idx - c * NPAD;
    float v = 0.f;
    if (p < NPIX) {
      int ly = p / TILE, lx = p - ly * TILE;
      int gy = r0 - 1 + ly, gx = c0 - 1 + lx;
      if ((unsigned)gy < IMG && (unsigned)gx < IMG)
        v = x[((size_t)(bi * DIM + c) << 16) + (gy << 8) + gx];
    }
    xs[p * DIM + c] = (__bf16)v;
  }
#endif
  __syncthreads();

  // ---- phase 1: conv1x1 192->576 as WMMA GEMM over the halo tile ----
  for (int mt = wave; mt < C3 / 16; mt += 8) {
    v16bf a[6];
#pragma unroll
    for (int kt = 0; kt < 6; ++kt) a[kt] = a_frag_global_bf16(w1bf, DIM, mt * 16, kt * 32);
    float bias[8];
#pragma unroll
    for (int r = 0; r < 8; ++r) bias[r] = b1[mt * 16 + r + half8];
#pragma unroll
    for (int nt = 0; nt < 7; ++nt) {
      v8f acc = {};
#pragma unroll
      for (int kt = 0; kt < 6; ++kt)
        acc = wmma_bf16(a[kt], b_frag_ldsT(xs, DIM, kt * 32, nt * 16), acc);
      int n = nt * 16 + ln;
#pragma unroll
      for (int r = 0; r < 8; ++r)
        ys[(mt * 16 + r + half8) * NPAD + n] = (__bf16)(acc[r] + bias[r]);
    }
  }
  __syncthreads();

  // ---- phase 2a: stage depthwise weights in LDS (xs region is dead) ----
  float* wdl = (float*)smem;            // 576*9 f32
  float* bdl = (float*)(smem + 20736);  // 576 f32
  for (int i = tid; i < C3 * 9; i += 256) wdl[i] = wd[i];
  for (int i = tid; i < C3; i += 256) bdl[i] = bd[i];
  __syncthreads();

  // ---- phase 2b: depthwise 3x3 on central 8x8, split q/k/v, apply temperature ----
  for (int idx = tid; idx < C3 * TOK; idx += 256) {
    int m = idx >> 6, tok = idx & 63;
    int ty = tok >> 3, tx = tok & 7;
    float s = bdl[m];
    const float*  wm   = wdl + m * 9;
    const __bf16* yrow = ys + m * NPAD;
#pragma unroll
    for (int dy = 0; dy < 3; ++dy)
#pragma unroll
      for (int dx = 0; dx < 3; ++dx)
        s += wm[dy * 3 + dx] * (float)yrow[(ty + dy) * TILE + (tx + dx)];
    if (m < DIM)            qs[tok * DIM + m]             = (__bf16)(s * temp[m >> 5]);
    else if (m < 2 * DIM)   ks[tok * DIM + (m - DIM)]     = (__bf16)s;
    else                    vs[(m - 2 * DIM) * TOK + tok] = (__bf16)s;
  }
  __syncthreads();

  // ---- phase 3: window attention; each wave owns (head, 16-token row block) ----
  __bf16* pw = probs + wave * 16 * 64;
  for (int rb = wave; rb < HEADS * 4; rb += 8) {
    int h = rb >> 2, mt = rb & 3;
    v16bf aq = a_frag_lds(qs + h * DH, DIM, mt * 16, 0);
    v8f s[4];
#pragma unroll
    for (int nt = 0; nt < 4; ++nt) {
      v8f z = {};
      s[nt] = wmma_bf16(aq, b_frag_ldsT(ks + h * DH, DIM, 0, nt * 16), z);
    }
    // softmax over 64 keys; a token row lives in one 16-lane half of the wave
#pragma unroll
    for (int r = 0; r < 8; ++r) {
      float mx = s[0][r];
#pragma unroll
      for (int nt = 1; nt < 4; ++nt) mx = fmaxf(mx, s[nt][r]);
#pragma unroll
      for (int d = 8; d >= 1; d >>= 1) mx = fmaxf(mx, __shfl_xor(mx, d, 32));
      float e[4], sum = 0.f;
#pragma unroll
      for (int nt = 0; nt < 4; ++nt) { e[nt] = __expf(s[nt][r] - mx); sum += e[nt]; }
#pragma unroll
      for (int d = 8; d >= 1; d >>= 1) sum += __shfl_xor(sum, d, 32);
      float inv = 1.f / sum;
      int prow = r + half8;
#pragma unroll
      for (int nt = 0; nt < 4; ++nt)
        pw[prow * 64 + nt * 16 + ln] = (__bf16)(e[nt] * inv);
    }
    // out = probs(16x64) @ v(64x32)
    v16bf ap0 = a_frag_lds(pw, 64, 0, 0);
    v16bf ap1 = a_frag_lds(pw, 64, 0, 32);
#pragma unroll
    for (int nt = 0; nt < 2; ++nt) {
      v8f o = {};
      o = wmma_bf16(ap0, b_frag_ldsT(vs + h * DH * TOK, TOK, 0,  nt * 16), o);
      o = wmma_bf16(ap1, b_frag_ldsT(vs + h * DH * TOK, TOK, 32, nt * 16), o);
#pragma unroll
      for (int r = 0; r < 8; ++r)
        ao[(mt * 16 + r + half8) * DIM + h * DH + nt * 16 + ln] = (__bf16)o[r];
    }
  }
  __syncthreads();

  // ---- phase 4: projection GEMM 192x192 + bias + residual -> global ----
  for (int t = wave; t < 48; t += 8) {
    int mt = t >> 2, nt = t & 3;
    v8f acc = {};
#pragma unroll
    for (int kt = 0; kt < 6; ++kt)
      acc = wmma_bf16(a_frag_global_bf16(wpbf, DIM, mt * 16, kt * 32),
                      b_frag_ldsT(ao, DIM, kt * 32, nt * 16), acc);
    int tok = nt * 16 + ln;
    int gy = r0 + (tok >> 3), gx = c0 + (tok & 7);
#pragma unroll
    for (int r = 0; r < 8; ++r) {
      int m = mt * 16 + r + half8;
      size_t o = ((size_t)(bi * DIM + m) << 16) + (gy << 8) + gx;
      out[o] = acc[r] + bp[m] + x[o];
    }
  }
}

extern "C" void kernel_launch(void* const* d_in, const int* in_sizes, int n_in,
                              void* d_out, int out_size, void* d_ws, size_t ws_size,
                              hipStream_t stream) {
  (void)n_in; (void)out_size; (void)ws_size;
  const float* x  = (const float*)d_in[0];
  const float* w1 = (const float*)d_in[1];
  const float* b1 = (const float*)d_in[2];
  const float* wd = (const float*)d_in[3];
  const float* bd = (const float*)d_in[4];
  const float* wp = (const float*)d_in[5];
  const float* bp = (const float*)d_in[6];
  const float* tp = (const float*)d_in[7];

  __bf16* wsbf = (__bf16*)d_ws;                 // [576*192] w1 | [192*192] wp, bf16
  const int ncvt = C3 * DIM + DIM * DIM;
  hipLaunchKernelGGL(cvt_weights, dim3((ncvt + 255) / 256), dim3(256), 0, stream,
                     w1, wp, wsbf);

  int bsz = in_sizes[0] / (DIM * IMG * IMG);
  dim3 grid((unsigned)(bsz * 32 * 32));
  size_t shmem = 245760;  // 240 KB of the 320 KB/WGP LDS
  hipLaunchKernelGGL(fused_window_attn, grid, dim3(256), shmem, stream,
                     x, wsbf, b1, wd, bd, wsbf + C3 * DIM, bp, tp, (float*)d_out);
}
